// Gemma3Attention_1022202217319
// MI455X (gfx1250) — compile-verified, matrix operand order checked by reading, not run on hardware
//
#include <hip/hip_runtime.h>
#include <cstdint>
#include <cstddef>

// ---------------- problem constants ----------------
#define B_      2
#define S_      2048
#define D_      2048
#define H_      16
#define KV_     4
#define HD_     128
#define WINDOW_ 1024
#define GROUPS_ (H_ / KV_)
#define M_      (B_ * S_)          // 4096 rows of tokens

#define SOFTCAP_ 50.0f
#define SCALING_ 0.0625f           // 256^-0.5
#define EPS_     1e-6f
#define NEG_INF_ (-2.3819763e+38f)

// ---------------- WMMA types (gfx1250, wave32) ----------------
typedef __bf16 bf16_t;
typedef bf16_t bf16x16 __attribute__((ext_vector_type(16)));
typedef float  f32x8   __attribute__((ext_vector_type(8)));

union FragB { bf16x16 v; uint4 q[2]; unsigned short u[16]; };
union FragF { f32x8 v; float f[8]; };

__device__ __forceinline__ unsigned short f2bf(float f) {
  union { float f; unsigned u; } x; x.f = f;
  unsigned r = x.u + 0x7FFFu + ((x.u >> 16) & 1u);   // round-to-nearest-even
  return (unsigned short)(r >> 16);
}

__device__ __forceinline__ f32x8 wmma_bf16(bf16x16 a, bf16x16 b, f32x8 c) {
  // 8 args: (neg_a, A, neg_b, B, c_mod, C, reuse_a, reuse_b)
  return __builtin_amdgcn_wmma_f32_16x16x32_bf16(false, a, false, b, (short)0, c,
                                                 false, false);
}

// Async global -> LDS 128-bit copy (CDNA5 TDM-lite path, tracked by ASYNCcnt).
// lds_off: wave-relative LDS byte offset (= low 32 bits of a generic LDS ptr).
__device__ __forceinline__ void async_g2l_b128(const void* gptr, unsigned lds_off) {
  asm volatile("global_load_async_to_lds_b128 %0, %1, off"
               :: "v"(lds_off), "v"((unsigned long long)(uintptr_t)gptr)
               : "memory");
}
__device__ __forceinline__ void wait_async0() {
  asm volatile("s_wait_asynccnt 0x0" ::: "memory");
}
__device__ __forceinline__ unsigned lds_off32(const void* p) {
  return (unsigned)(uintptr_t)p;   // generic LDS addr: [31:0] = LDS offset
}

// ---------------- f32 -> bf16 conversion (grid-stride) ----------------
__global__ __launch_bounds__(256)
void cvt_f32_bf16_kernel(const float* __restrict__ in,
                         unsigned short* __restrict__ out, long long n) {
  long long i = (long long)blockIdx.x * blockDim.x + threadIdx.x;
  long long stride = (long long)gridDim.x * blockDim.x;
  for (; i < n; i += stride) out[i] = f2bf(in[i]);
}

// ---------------- bf16 GEMM: C[M,N] = A[M,K] * W[N,K]^T (f32 accum) -----
// block tile 128x128, BK=32, 256 threads = 8 waves (2 x 4), wave tile 64x32.
// Double-buffered LDS, global->LDS via async copies (ASYNCcnt), one barrier
// per K-step: loads of tile i+1 overlap WMMA of tile i.
#define GBM 128
#define GBN 128
#define GBK 32
#define GPAD 40   // LDS row stride (halfwords): 32 data + 8 pad, 80B (16B aligned)

__global__ __launch_bounds__(256)
void gemm_bf16_kernel(const unsigned short* __restrict__ A,   // [M,K] bf16
                      const unsigned short* __restrict__ W,   // [N,K] bf16
                      float* __restrict__ C,                  // [M,N] f32
                      int M, int N, int K) {
  __shared__ __attribute__((aligned(16))) unsigned short sA[2][GBM * GPAD];
  __shared__ __attribute__((aligned(16))) unsigned short sB[2][GBN * GPAD];

  const int tid  = threadIdx.x;
  const int lane = tid & 31;
  const int wid  = tid >> 5;
  const int half = lane >> 4;     // 0: lanes 0-15, 1: lanes 16-31
  const int l16  = lane & 15;
  const int wm   = wid >> 2;      // 0..1  (64 rows each)
  const int wn   = wid & 3;       // 0..3  (32 cols each)
  const int m0   = blockIdx.x * GBM;
  const int n0   = blockIdx.y * GBN;

  // 512 16B chunks per tile per matrix; 2 chunks each for 256 threads.
  const int r_ld  = (tid + 0)   >> 2;            // chunk 0 row
  const int c8_ld = (tid & 3) * 8;               // chunk 0 col (halfwords)
  const int r_ld2  = (tid + 256) >> 2;           // chunk 1 row
  const int c8_ld2 = c8_ld;                      // same col pattern

  auto issue_tile = [&](int k0, int buf) {
    const unsigned short* ap0 = A + (size_t)(m0 + r_ld)  * K + k0 + c8_ld;
    const unsigned short* bp0 = W + (size_t)(n0 + r_ld)  * K + k0 + c8_ld;
    const unsigned short* ap1 = A + (size_t)(m0 + r_ld2) * K + k0 + c8_ld2;
    const unsigned short* bp1 = W + (size_t)(n0 + r_ld2) * K + k0 + c8_ld2;
    if (k0 + GBK < K) {                    // prefetch tile-after-next into GL2
      __builtin_prefetch(ap0 + GBK, 0, 1);
      __builtin_prefetch(bp0 + GBK, 0, 1);
    }
    async_g2l_b128(ap0, lds_off32(&sA[buf][r_ld  * GPAD + c8_ld]));
    async_g2l_b128(bp0, lds_off32(&sB[buf][r_ld  * GPAD + c8_ld]));
    async_g2l_b128(ap1, lds_off32(&sA[buf][r_ld2 * GPAD + c8_ld2]));
    async_g2l_b128(bp1, lds_off32(&sB[buf][r_ld2 * GPAD + c8_ld2]));
  };

  FragF acc[4][2];
  for (int mi = 0; mi < 4; ++mi)
    for (int ni = 0; ni < 2; ++ni)
      for (int r = 0; r < 8; ++r) acc[mi][ni].f[r] = 0.0f;

  const int nk = K / GBK;
  issue_tile(0, 0);

  for (int i = 0; i < nk; ++i) {
    const int buf = i & 1;
    wait_async0();           // this wave's async copies for tile i landed
    __syncthreads();         // everyone's copies landed; prev readers done
    if (i + 1 < nk) issue_tile((i + 1) * GBK, buf ^ 1);  // overlap with WMMA

    FragB af[4], bf[2];
    #pragma unroll
    for (int mi = 0; mi < 4; ++mi) {
      const unsigned short* p = &sA[buf][(wm * 64 + mi * 16 + l16) * GPAD + half * 8];
      af[mi].q[0] = *(const uint4*)p;          // K 0..7   (+8 for high half)
      af[mi].q[1] = *(const uint4*)(p + 16);   // K 16..23 (+8 for high half)
    }
    #pragma unroll
    for (int ni = 0; ni < 2; ++ni) {
      const unsigned short* p = &sB[buf][(wn * 32 + ni * 16 + l16) * GPAD + half * 8];
      bf[ni].q[0] = *(const uint4*)p;
      bf[ni].q[1] = *(const uint4*)(p + 16);
    }
    #pragma unroll
    for (int mi = 0; mi < 4; ++mi)
      #pragma unroll
      for (int ni = 0; ni < 2; ++ni)
        acc[mi][ni].v = wmma_bf16(af[mi].v, bf[ni].v, acc[mi][ni].v);
  }

  // C fragment layout: VGPR r -> M = r + 8*half, N = lane&15
  #pragma unroll
  for (int mi = 0; mi < 4; ++mi)
    #pragma unroll
    for (int ni = 0; ni < 2; ++ni)
      #pragma unroll
      for (int r = 0; r < 8; ++r) {
        int row = m0 + wm * 64 + mi * 16 + r + 8 * half;
        int col = n0 + wn * 32 + ni * 16 + l16;
        C[(size_t)row * N + col] = acc[mi][ni].f[r];
      }
}

// ---------------- RMS-norm + RoPE for Q and K (one wave per head-row) ---
__global__ __launch_bounds__(256)
void qk_norm_rope_kernel(const float* __restrict__ qraw,   // [B,S,H,HD]
                         const float* __restrict__ kraw,   // [B,S,KV,HD]
                         const float* __restrict__ cosp,   // [S,HD]
                         const float* __restrict__ sinp,   // [S,HD]
                         const float* __restrict__ qw,     // [HD]
                         const float* __restrict__ kw,     // [HD]
                         unsigned short* __restrict__ Q16, // [B,H,S,HD]
                         unsigned short* __restrict__ K16) // [B,KV,S,HD]
{
  const int nwaves = blockDim.x >> 5;
  int row = blockIdx.x * nwaves + (threadIdx.x >> 5);
  const int lane = threadIdx.x & 31;
  const int ROWS = B_ * S_ * (H_ + KV_);
  if (row >= ROWS) return;

  int hh = row % (H_ + KV_);
  int bs = row / (H_ + KV_);
  int s  = bs % S_;
  int b  = bs / S_;

  const float* src;
  const float* w;
  unsigned short* dst;
  if (hh < H_) {
    src = qraw + ((size_t)bs * H_ + hh) * HD_;
    w   = qw;
    dst = Q16 + (((size_t)(b * H_ + hh)) * S_ + s) * HD_;
  } else {
    int kh = hh - H_;
    src = kraw + ((size_t)bs * KV_ + kh) * HD_;
    w   = kw;
    dst = K16 + (((size_t)(b * KV_ + kh)) * S_ + s) * HD_;
  }

  // each lane owns d = lane, lane+32, lane+64, lane+96 -> rotate pairs local
  float x0 = src[lane], x1 = src[lane + 32], x2 = src[lane + 64], x3 = src[lane + 96];
  float ss = x0 * x0 + x1 * x1 + x2 * x2 + x3 * x3;
  #pragma unroll
  for (int off = 1; off < 32; off <<= 1) ss += __shfl_xor(ss, off, 32);
  float inv = rsqrtf(ss * (1.0f / HD_) + EPS_);

  float n0 = x0 * inv * (1.0f + w[lane]);
  float n1 = x1 * inv * (1.0f + w[lane + 32]);
  float n2 = x2 * inv * (1.0f + w[lane + 64]);
  float n3 = x3 * inv * (1.0f + w[lane + 96]);

  const float* cs = cosp + (size_t)s * HD_;
  const float* sn = sinp + (size_t)s * HD_;
  float y0 = n0 * cs[lane]      + (-n2) * sn[lane];
  float y1 = n1 * cs[lane + 32] + (-n3) * sn[lane + 32];
  float y2 = n2 * cs[lane + 64] + ( n0) * sn[lane + 64];
  float y3 = n3 * cs[lane + 96] + ( n1) * sn[lane + 96];

  dst[lane]      = f2bf(y0);
  dst[lane + 32] = f2bf(y1);
  dst[lane + 64] = f2bf(y2);
  dst[lane + 96] = f2bf(y3);
}

// ---------------- V repack: f32 [B,S,KV,HD] -> bf16 [B,KV,S,HD] ---------
__global__ __launch_bounds__(256)
void v_pack_kernel(const float* __restrict__ vraw, unsigned short* __restrict__ V16) {
  const long long N = (long long)B_ * S_ * KV_ * HD_;
  long long i = (long long)blockIdx.x * blockDim.x + threadIdx.x;
  long long stride = (long long)gridDim.x * blockDim.x;
  for (; i < N; i += stride) {
    int d  = (int)(i & (HD_ - 1));
    long long t = i >> 7;
    int kv = (int)(t & (KV_ - 1));
    t >>= 2;
    int s = (int)(t & (S_ - 1));
    int b = (int)(t >> 11);
    V16[(((size_t)(b * KV_ + kv)) * S_ + s) * HD_ + d] = f2bf(vraw[i]);
  }
}

// ---------------- sliding-window flash attention ------------------------
// grid = (S/64, B*H); 128 threads = 4 waves; 16 queries per wave; 32 keys/iter.
// K tile staged with async global->LDS copies; V tile transposed via VGPRs.
#define KPAD 136   // sK row stride (halfwords) for 128 d-elements
#define VPAD 40    // sVT row stride (halfwords) for 32 keys
#define PPAD 40    // per-wave P tile row stride

__global__ __launch_bounds__(128)
void attn_kernel(const unsigned short* __restrict__ Q16,  // [B,H,S,HD]
                 const unsigned short* __restrict__ K16,  // [B,KV,S,HD]
                 const unsigned short* __restrict__ V16,  // [B,KV,S,HD]
                 unsigned short* __restrict__ AO)         // [B,S,H*HD]
{
  __shared__ __attribute__((aligned(16))) unsigned short sK[32 * KPAD];
  __shared__ __attribute__((aligned(16))) unsigned short sVT[HD_ * VPAD];
  __shared__ __attribute__((aligned(16))) unsigned short sP[4 * 16 * PPAD];

  const int tid  = threadIdx.x;
  const int lane = tid & 31;
  const int wid  = tid >> 5;
  const int half = lane >> 4;
  const int l16  = lane & 15;

  const int b  = blockIdx.y / H_;
  const int h  = blockIdx.y % H_;
  const int kv = h / GROUPS_;
  const int q0blk = blockIdx.x * 64;
  const int qw    = q0blk + wid * 16;   // this wave's first query row

  const unsigned short* Qp = Q16 + (((size_t)(b * H_ + h)) * S_ + qw) * HD_;
  const unsigned short* Kp = K16 + ((size_t)(b * KV_ + kv)) * S_ * HD_;
  const unsigned short* Vp = V16 + ((size_t)(b * KV_ + kv)) * S_ * HD_;

  // Q fragments: 4 chunks of K-dim 32 over HD=128, resident for whole loop
  FragB qf[4];
  #pragma unroll
  for (int c = 0; c < 4; ++c) {
    const unsigned short* p = Qp + (size_t)l16 * HD_ + c * 32 + half * 8;
    qf[c].q[0] = *(const uint4*)p;
    qf[c].q[1] = *(const uint4*)(p + 16);
  }

  float m[8], lsum[8];
  FragF O[8];
  #pragma unroll
  for (int r = 0; r < 8; ++r) { m[r] = NEG_INF_; lsum[r] = 0.0f; }
  #pragma unroll
  for (int g = 0; g < 8; ++g)
    #pragma unroll
    for (int r = 0; r < 8; ++r) O[g].f[r] = 0.0f;

  int kstart = q0blk - (WINDOW_ - 1);
  if (kstart < 0) kstart = 0;
  kstart &= ~31;
  const int kend = q0blk + 63;

  for (int kt0 = kstart; kt0 <= kend; kt0 += 32) {
    // ---- stage K tile [32][128] (async -> LDS) and V^T tile [128][32]
    #pragma unroll
    for (int t = 0; t < 4; ++t) {
      int c   = tid + t * 128;
      int key = c >> 4;
      int d8  = (c & 15) * 8;
      const unsigned short* kp = Kp + (size_t)(kt0 + key) * HD_ + d8;
      async_g2l_b128(kp, lds_off32(&sK[key * KPAD + d8]));
      union { uint4 q; unsigned short u[8]; } vu;
      vu.q = *(const uint4*)(Vp + (size_t)(kt0 + key) * HD_ + d8);
      #pragma unroll
      for (int i = 0; i < 8; ++i) sVT[(d8 + i) * VPAD + key] = vu.u[i];
    }
    wait_async0();       // this wave's K copies done (V ds-stores via barrier)
    __syncthreads();

    // wave-uniform activity test against this wave's window
    const bool active = (kt0 <= qw + 15) && (kt0 + 31 >= qw - (WINDOW_ - 1));
    if (active) {
      FragF sc[2];
      #pragma unroll
      for (int kt = 0; kt < 2; ++kt) {
        #pragma unroll
        for (int r = 0; r < 8; ++r) sc[kt].f[r] = 0.0f;
        #pragma unroll
        for (int c = 0; c < 4; ++c) {
          FragB kf;
          const unsigned short* p = &sK[(kt * 16 + l16) * KPAD + c * 32 + half * 8];
          kf.q[0] = *(const uint4*)p;
          kf.q[1] = *(const uint4*)(p + 16);
          sc[kt].v = wmma_bf16(qf[c].v, kf.v, sc[kt].v);
        }
        // scale -> softcap -> window mask (C layout: row = r+8*half, col = l16)
        #pragma unroll
        for (int r = 0; r < 8; ++r) {
          float x = sc[kt].f[r] * SCALING_;
          x = SOFTCAP_ * tanhf(x * (1.0f / SOFTCAP_));
          int qi = qw + r + 8 * half;
          int ki = kt0 + kt * 16 + l16;
          if (!((ki <= qi) && (qi - ki < WINDOW_))) x = NEG_INF_;
          sc[kt].f[r] = x;
        }
      }

      // ---- online softmax (row stats live per accumulator slot r)
      float alpha[8];
      #pragma unroll
      for (int r = 0; r < 8; ++r) {
        float v = fmaxf(sc[0].f[r], sc[1].f[r]);
        #pragma unroll
        for (int off = 1; off < 16; off <<= 1) v = fmaxf(v, __shfl_xor(v, off, 32));
        float mn = fmaxf(m[r], v);
        alpha[r] = __expf(m[r] - mn);
        m[r] = mn;
        float p0 = __expf(sc[0].f[r] - mn);
        float p1 = __expf(sc[1].f[r] - mn);
        sc[0].f[r] = p0;
        sc[1].f[r] = p1;
        float rs = p0 + p1;
        #pragma unroll
        for (int off = 1; off < 16; off <<= 1) rs += __shfl_xor(rs, off, 32);
        lsum[r] = lsum[r] * alpha[r] + rs;
      }
      #pragma unroll
      for (int g = 0; g < 8; ++g)
        #pragma unroll
        for (int r = 0; r < 8; ++r) O[g].f[r] *= alpha[r];

      // ---- C-layout P -> A-layout via per-wave LDS round trip
      unsigned short* pw = &sP[wid * 16 * PPAD];
      #pragma unroll
      for (int kt = 0; kt < 2; ++kt)
        #pragma unroll
        for (int r = 0; r < 8; ++r)
          pw[(r + 8 * half) * PPAD + kt * 16 + l16] = f2bf(sc[kt].f[r]);
      asm volatile("s_wait_dscnt 0x0" ::: "memory");
      __builtin_amdgcn_wave_barrier();

      FragB pf;
      {
        const unsigned short* p = &pw[l16 * PPAD + half * 8];
        pf.q[0] = *(const uint4*)p;
        pf.q[1] = *(const uint4*)(p + 16);
      }

      // ---- O += P(16x32) x V(32x128) as 8 WMMAs against transposed V
      #pragma unroll
      for (int g = 0; g < 8; ++g) {
        FragB vf;
        const unsigned short* p = &sVT[(g * 16 + l16) * VPAD + half * 8];
        vf.q[0] = *(const uint4*)p;
        vf.q[1] = *(const uint4*)(p + 16);
        O[g].v = wmma_bf16(pf.v, vf.v, O[g].v);
      }
    }
    __syncthreads();
  }

  // ---- epilogue: normalize and store bf16 [B,S,H*HD]
  #pragma unroll
  for (int g = 0; g < 8; ++g)
    #pragma unroll
    for (int r = 0; r < 8; ++r) {
      float val = O[g].f[r] / lsum[r];
      int row = qw + r + 8 * half;
      int col = g * 16 + l16;
      AO[((size_t)(b * S_ + row)) * (H_ * HD_) + h * HD_ + col] = f2bf(val);
    }
}

// ---------------- launcher ----------------------------------------------
extern "C" void kernel_launch(void* const* d_in, const int* in_sizes, int n_in,
                              void* d_out, int out_size, void* d_ws, size_t ws_size,
                              hipStream_t stream) {
  (void)in_sizes; (void)n_in; (void)out_size; (void)ws_size;

  const float* hidden = (const float*)d_in[0];   // [B,S,D]
  const float* cosp   = (const float*)d_in[1];   // [1,S,HD]
  const float* sinp   = (const float*)d_in[2];   // [1,S,HD]
  const float* wq     = (const float*)d_in[3];   // [H*HD, D]
  const float* wk     = (const float*)d_in[4];   // [KV*HD, D]
  const float* wv     = (const float*)d_in[5];   // [KV*HD, D]
  const float* wo     = (const float*)d_in[6];   // [D, H*HD]
  const float* qnw    = (const float*)d_in[7];   // [HD]
  const float* knw    = (const float*)d_in[8];   // [HD]
  float* out = (float*)d_out;                    // [B,S,D] f32

  // ---- workspace carve (256B aligned)
  char* p = (char*)d_ws;
  auto carve = [&](size_t bytes) {
    void* r = (void*)p;
    p += (bytes + 255) & ~(size_t)255;
    return r;
  };
  unsigned short* X16  = (unsigned short*)carve((size_t)M_ * D_ * 2);          // 16 MB
  unsigned short* Wq16 = (unsigned short*)carve((size_t)H_ * HD_ * D_ * 2);    // 8 MB
  unsigned short* Wk16 = (unsigned short*)carve((size_t)KV_ * HD_ * D_ * 2);   // 2 MB
  unsigned short* Wv16 = (unsigned short*)carve((size_t)KV_ * HD_ * D_ * 2);   // 2 MB
  unsigned short* Wo16 = (unsigned short*)carve((size_t)D_ * H_ * HD_ * 2);    // 8 MB
  float* qraw = (float*)carve((size_t)M_ * H_ * HD_ * 4);                      // 33.5 MB
  float* kraw = (float*)carve((size_t)M_ * KV_ * HD_ * 4);                     // 8.4 MB
  float* vraw = (float*)carve((size_t)M_ * KV_ * HD_ * 4);                     // 8.4 MB
  unsigned short* Q16 = (unsigned short*)carve((size_t)B_ * H_ * S_ * HD_ * 2);  // 16 MB
  unsigned short* K16 = (unsigned short*)carve((size_t)B_ * KV_ * S_ * HD_ * 2); // 4 MB
  unsigned short* V16 = (unsigned short*)carve((size_t)B_ * KV_ * S_ * HD_ * 2); // 4 MB
  unsigned short* AO16 = (unsigned short*)qraw;  // overlay: qraw dead after rope

  // 1) f32 -> bf16 conversions
  cvt_f32_bf16_kernel<<<2048, 256, 0, stream>>>(hidden, X16, (long long)M_ * D_);
  cvt_f32_bf16_kernel<<<1024, 256, 0, stream>>>(wq, Wq16, (long long)H_ * HD_ * D_);
  cvt_f32_bf16_kernel<<<512,  256, 0, stream>>>(wk, Wk16, (long long)KV_ * HD_ * D_);
  cvt_f32_bf16_kernel<<<512,  256, 0, stream>>>(wv, Wv16, (long long)KV_ * HD_ * D_);
  cvt_f32_bf16_kernel<<<1024, 256, 0, stream>>>(wo, Wo16, (long long)D_ * H_ * HD_);

  // 2) QKV projections (WMMA GEMMs)
  gemm_bf16_kernel<<<dim3(M_ / GBM, (H_ * HD_) / GBN), 256, 0, stream>>>(
      X16, Wq16, qraw, M_, H_ * HD_, D_);
  gemm_bf16_kernel<<<dim3(M_ / GBM, (KV_ * HD_) / GBN), 256, 0, stream>>>(
      X16, Wk16, kraw, M_, KV_ * HD_, D_);
  gemm_bf16_kernel<<<dim3(M_ / GBM, (KV_ * HD_) / GBN), 256, 0, stream>>>(
      X16, Wv16, vraw, M_, KV_ * HD_, D_);

  // 3) RMS-norm + RoPE for Q,K; repack V  (head-major bf16)
  {
    int rows = B_ * S_ * (H_ + KV_);
    int blocks = (rows + 7) / 8;   // 8 waves per 256-thread block
    qk_norm_rope_kernel<<<blocks, 256, 0, stream>>>(qraw, kraw, cosp, sinp,
                                                    qnw, knw, Q16, K16);
  }
  v_pack_kernel<<<2048, 256, 0, stream>>>(vraw, V16);

  // 4) sliding-window flash attention (WMMA)
  attn_kernel<<<dim3(S_ / 64, B_ * H_), 128, 0, stream>>>(Q16, K16, V16, AO16);

  // 5) output projection (WMMA GEMM) -> f32 d_out
  gemm_bf16_kernel<<<dim3(M_ / GBM, D_ / GBN), 256, 0, stream>>>(
      AO16, Wo16, out, M_, D_, H_ * HD_);
}
